// WaveRNN_12962211299600
// MI455X (gfx1250) — matrix operand order repaired and checked by reading, not compiled, not run on hardware
//
#include <hip/hip_runtime.h>
#include <hip/hip_bf16.h>
#include <math.h>

// ---------------- problem constants (from reference) ----------------
#define B_   4
#define L_   48
#define F_   80
#define C_   256
#define H_   896
#define H3_  2688          // 3*H
#define A_   256
#define UP_  4
#define REP_ 75
#define SEG_ 192           // L*UP distinct cond columns
#define T_   14400         // L*UP*REPEAT
#define ROWS_ (T_*B_)      // 57600 (t-major, row = t*B + b)

// ---------------- persistent GRU kernel geometry ----------------
#define NWG_ 16            // one cluster: 16 workgroups, one per WGP
#define CPW_ 168           // columns of 3H owned per workgroup (16*168 = 2688)
#define NT_  11            // 16-wide column tiles per WG (tile 10 half-padded)
#define KT_  28            // 896 / 32 K-chunks
#define IPW_ 56            // rows of h updated per WG (16*56 = 896)
#define MPAD_ 16           // padded M rows for the h staging buffer

typedef float  v8f   __attribute__((ext_vector_type(8)));
typedef __bf16 v16bf __attribute__((ext_vector_type(16)));
typedef __bf16 v8bf  __attribute__((ext_vector_type(8)));

// =====================================================================
// Small prep kernels (negligible FLOPs; keep them simple VALU code)
// =====================================================================

// x1[b][c][l] = dot(local_features[b][l][:F], proj_w[c][:F]) + proj_b[c]
__global__ void kProj(const float* __restrict__ lf, const float* __restrict__ pw,
                      const float* __restrict__ pb, float* __restrict__ x1)
{
    int idx = blockIdx.x * blockDim.x + threadIdx.x;           // B*C*L
    if (idx >= B_ * C_ * L_) return;
    int l = idx % L_, c = (idx / L_) % C_, b = idx / (L_ * C_);
    const float* a = lf + ((size_t)b * L_ + l) * F_;
    const float* w = pw + (size_t)c * F_;
    float s = pb[c];
    for (int f = 0; f < F_; ++f) s += a[f] * w[f];
    x1[idx] = s;
}

// cond[seg][b][o] = sum_c x1[b][c][l]*up_w[o][c][k] + up_b[o],  seg = 4l+k
__global__ void kUp(const float* __restrict__ x1, const float* __restrict__ upw,
                    const float* __restrict__ upb, float* __restrict__ cond)
{
    int idx = blockIdx.x * blockDim.x + threadIdx.x;           // SEG*B*C
    if (idx >= SEG_ * B_ * C_) return;
    int o = idx % C_, b = (idx / C_) % B_, seg = idx / (C_ * B_);
    int l = seg >> 2, k = seg & 3;
    float s = upb[o];
    for (int c = 0; c < C_; ++c)
        s += x1[((size_t)b * C_ + c) * L_ + l] * upw[((size_t)o * C_ + c) * UP_ + k];
    cond[idx] = s;
}

// v[j] = Wi[j]·embed_w ; cvec[j] = Wi[j]·embed_b + gru_bi[j]
__global__ void kVC(const float* __restrict__ wi, const float* __restrict__ ew,
                    const float* __restrict__ eb, const float* __restrict__ bi,
                    float* __restrict__ v, float* __restrict__ cvec)
{
    int j = blockIdx.x * blockDim.x + threadIdx.x;
    if (j >= H3_) return;
    const float* w = wi + (size_t)j * C_;
    float sv = 0.f, sc = 0.f;
    for (int c = 0; c < C_; ++c) { sv += w[c] * ew[c]; sc += w[c] * eb[c]; }
    v[j] = sv; cvec[j] = sc + bi[j];
}

// U[seg][b][j] = sum_c cond[seg][b][c] * Wi[j][c]
__global__ void kCondProj(const float* __restrict__ cond, const float* __restrict__ wi,
                          float* __restrict__ U)
{
    int idx = blockIdx.x * blockDim.x + threadIdx.x;           // SEG*B*H3
    if (idx >= SEG_ * B_ * H3_) return;
    int j = idx % H3_;
    int rb = idx / H3_;                                        // seg*B+b
    const float* cc = cond + (size_t)rb * C_;
    const float* w  = wi + (size_t)j * C_;
    float s = 0.f;
    for (int c = 0; c < C_; ++c) s += cc[c] * w[c];
    U[idx] = s;
}

// Pack gru_wh (fp32 [3H][H]) into per-lane bf16 WMMA B-fragments:
// whp[((wg*NT + nt)*KT + kt)*32 + lane][0..15] ; B[k][n]=Wh[colBase+n][kBase+k]
// lane: n = lane&15, k = (lane>>4)*16 + e. Zero-pad beyond 168 local cols.
__global__ void kPackWh(const float* __restrict__ wh, __bf16* __restrict__ whp)
{
    int idx = blockIdx.x * blockDim.x + threadIdx.x;           // NWG*NT*KT*32
    if (idx >= NWG_ * NT_ * KT_ * 32) return;
    int lane = idx & 31, rest = idx >> 5;
    int kt = rest % KT_; rest /= KT_;
    int nt = rest % NT_; int wg = rest / NT_;
    int lcol = nt * 16 + (lane & 15);
    int kbase = kt * 32 + (lane >> 4) * 16;
    __bf16* dst = whp + (size_t)idx * 16;
    if (lcol < CPW_) {
        const float* src = wh + (size_t)(wg * CPW_ + lcol) * H_ + kbase;
        #pragma unroll
        for (int e = 0; e < 16; ++e) dst[e] = (__bf16)src[e];
    } else {
        #pragma unroll
        for (int e = 0; e < 16; ++e) dst[e] = (__bf16)0.0f;
    }
}

__global__ void kCvtBf16(const float* __restrict__ src, __bf16* __restrict__ dst, int n)
{
    int i = blockIdx.x * blockDim.x + threadIdx.x;
    if (i < n) dst[i] = (__bf16)src[i];
}

// hbf has MPAD_ (16) rows; rows 4..15 stay zero forever so A-fragment loads
// in the GRU hot loop are branchless.
__global__ void kZeroState(float* __restrict__ hf, __bf16* __restrict__ hbf,
                           unsigned* __restrict__ bar)
{
    for (int i = threadIdx.x; i < B_ * H_; i += blockDim.x) hf[i] = 0.f;
    for (int i = threadIdx.x; i < MPAD_ * H_; i += blockDim.x) hbf[i] = (__bf16)0.0f;
    if (threadIdx.x < 2) bar[threadIdx.x] = 0u;
}

// =====================================================================
// Fragment helpers (static register usage only — no arrays)
// =====================================================================

// A-matrix 16x32 bf16: lane<16 -> M=lane, K = 0..15;
// lane>=16 -> M=lane-16, K = 16..31. src has 16 zero-padded rows,
// so the load is unconditional (no EXEC juggling in the hot loop).
__device__ __forceinline__ v16bf loadAfrag(const __bf16* __restrict__ src,
                                           int row, int khalf, int kbase)
{
    const v8bf lo = *(const v8bf*)(src + (size_t)row * H_ + kbase + khalf);
    const v8bf hi = *(const v8bf*)(src + (size_t)row * H_ + kbase + khalf + 8);
    return __builtin_shufflevector(lo, hi, 0, 1, 2, 3, 4, 5, 6, 7,
                                           8, 9, 10, 11, 12, 13, 14, 15);
}

__device__ __forceinline__ v16bf loadBfragLds(const __bf16* lds, int nt, int kt, int lane)
{
    return *(const v16bf*)(lds + ((size_t)(nt * KT_ + kt) * 32 + lane) * 16);
}

// D layout: value v in lane -> M = v + (lane>>4)*8, N = lane&15.
// Batch rows (M<4) live in VGPR b of lanes 0..15.
__device__ __forceinline__ void storeGh(float* __restrict__ gh, v8f acc,
                                        int wg, int nt, int lane)
{
    const int lcol = nt * 16 + (lane & 15);
    if (lane < 16 && lcol < CPW_) {
        const int col = wg * CPW_ + lcol;
        #pragma unroll
        for (int b = 0; b < B_; ++b) gh[b * H3_ + col] = acc[b];
    }
}

// =====================================================================
// Grid-wide sync: HW cluster barrier when launched as one 16-WG cluster,
// global-memory sense-reversing barrier otherwise.
// =====================================================================
__device__ __forceinline__ void gridSync(unsigned* cnt, unsigned* phase,
                                         unsigned nBlocks, bool inCluster)
{
    __threadfence();
    __syncthreads();
    if (inCluster) {
        __builtin_amdgcn_s_cluster_barrier();                  // s_barrier_signal/wait -3
    } else {
        if (threadIdx.x == 0) {
            unsigned my  = __hip_atomic_load(phase, __ATOMIC_RELAXED, __HIP_MEMORY_SCOPE_AGENT);
            unsigned arr = __hip_atomic_fetch_add(cnt, 1u, __ATOMIC_ACQ_REL, __HIP_MEMORY_SCOPE_AGENT);
            if (arr == nBlocks - 1u) {
                __hip_atomic_store(cnt, 0u, __ATOMIC_RELAXED, __HIP_MEMORY_SCOPE_AGENT);
                __hip_atomic_store(phase, my + 1u, __ATOMIC_RELEASE, __HIP_MEMORY_SCOPE_AGENT);
            } else {
                while (__hip_atomic_load(phase, __ATOMIC_ACQUIRE, __HIP_MEMORY_SCOPE_AGENT) == my)
                    __builtin_amdgcn_s_sleep(1);
            }
        }
        __syncthreads();
    }
    __threadfence();
}

// =====================================================================
// Persistent GRU: 16 blocks x 128 threads (4 waves). Each block keeps its
// 168-column slice of Wh as bf16 WMMA fragments in LDS (~308 KB of the
// 320 KB WGP LDS) and performs h@Wh^T with v_wmma_f32_16x16x32_bf16.
// Tiles per wave: wave, wave+4, min(wave+8,10). Wave 3 redundantly
// recomputes tile 10 (identical operands/result) so the hot loop has no
// divergent branch and no EXEC save/restore around WMMA. The K-loop is
// software-pipelined: next chunk's A/B fragments are fetched while the
// current three WMMAs execute.
// =====================================================================
__global__ void __launch_bounds__(128) kGru(
    const __bf16* __restrict__ whp, const float* __restrict__ U,
    const float* __restrict__ vvec, const float* __restrict__ cvec,
    const float* __restrict__ gold, const float* __restrict__ bh,
    float* __restrict__ gh, float* __restrict__ hf,
    __bf16* __restrict__ hbf, __bf16* __restrict__ hs,
    unsigned* __restrict__ barCnt, unsigned* __restrict__ barPhase)
{
    extern __shared__ __align__(16) unsigned char smemRaw[];
    __bf16* lds = (__bf16*)smemRaw;

    const int tid = threadIdx.x, wg = blockIdx.x;
    const int lane = tid & 31, wave = tid >> 5;
    const bool inCluster = (__builtin_amdgcn_cluster_id_x() != 0);

    // Stage this WG's Wh fragments into LDS (NT*KT*32 lane-fragments of 32B).
    const size_t frags = (size_t)NT_ * KT_ * 32;
    const __bf16* gsrc = whp + (size_t)wg * frags * 16;
    for (size_t f = tid; f < frags; f += 128) {
        const int4* s = (const int4*)(gsrc + f * 16);
        int4* d = (int4*)(lds + f * 16);
        d[0] = s[0]; d[1] = s[1];
    }
    __syncthreads();

    const v8f vzero = {0.f,0.f,0.f,0.f,0.f,0.f,0.f,0.f};
    const int row   = lane & 15;            // A-matrix M row (batch, padded to 16)
    const int khalf = (lane >> 4) * 16;     // contiguous K halves: lane half h takes K 16h..16h+15
    const int nt0 = wave, nt1 = wave + 4;
    const int nt2 = (wave + 8 < NT_) ? wave + 8 : NT_ - 1;     // wave3 dups tile 10

    for (int t = 0; t < T_; ++t) {
        // ---- matvec: gh[:, wg*168 .. +167] = h @ Wh^T  (WMMA bf16) ----
        v8f acc0 = vzero, acc1 = vzero, acc2 = vzero;
        v16bf afrag = loadAfrag(hbf, row, khalf, 0);
        v16bf bf0 = loadBfragLds(lds, nt0, 0, lane);
        v16bf bf1 = loadBfragLds(lds, nt1, 0, lane);
        v16bf bf2 = loadBfragLds(lds, nt2, 0, lane);
        for (int kt = 0; kt < KT_; ++kt) {
            const int ktn = (kt + 1 < KT_) ? kt + 1 : kt;      // prefetch next chunk
            const v16bf afragN = loadAfrag(hbf, row, khalf, ktn * 32);
            const v16bf bf0N = loadBfragLds(lds, nt0, ktn, lane);
            const v16bf bf1N = loadBfragLds(lds, nt1, ktn, lane);
            const v16bf bf2N = loadBfragLds(lds, nt2, ktn, lane);
            acc0 = __builtin_amdgcn_wmma_f32_16x16x32_bf16(
                false, afrag, false, bf0, (short)0, acc0, false, false);
            acc1 = __builtin_amdgcn_wmma_f32_16x16x32_bf16(
                false, afrag, false, bf1, (short)0, acc1, false, false);
            acc2 = __builtin_amdgcn_wmma_f32_16x16x32_bf16(
                false, afrag, false, bf2, (short)0, acc2, false, false);
            afrag = afragN; bf0 = bf0N; bf1 = bf1N; bf2 = bf2N;
        }
        storeGh(gh, acc0, wg, nt0, lane);
        storeGh(gh, acc1, wg, nt1, lane);
        storeGh(gh, acc2, wg, nt2, lane);   // wave3 rewrites tile 10 (same values)

        gridSync(barCnt, barPhase, NWG_, inCluster);

        // ---- pointwise GRU update for our 56 rows of h ----
        const int seg = t / REP_;
        if (((t + 1) % REP_) == 0 && (t + 1) < T_) {
            // warm L2 for next segment's input projections (global_prefetch)
            const float* nxt = U + ((size_t)(((t + 1) / REP_) * B_)) * H3_;
            __builtin_prefetch(nxt + (size_t)tid * 84, 0, 1);
        }
        for (int idx = tid; idx < B_ * IPW_; idx += 128) {
            const int b = idx / IPW_;
            const int i = wg * IPW_ + (idx % IPW_);
            const float g   = gold[(size_t)b * T_ + t];
            const float* Up = U + ((size_t)(seg * B_ + b)) * H3_;
            const float xr = Up[i]           + g * vvec[i]           + cvec[i];
            const float xz = Up[H_ + i]      + g * vvec[H_ + i]      + cvec[H_ + i];
            const float xn = Up[2 * H_ + i]  + g * vvec[2 * H_ + i]  + cvec[2 * H_ + i];
            const float hr = gh[b * H3_ + i]          + bh[i];
            const float hz = gh[b * H3_ + H_ + i]     + bh[H_ + i];
            const float hn = gh[b * H3_ + 2 * H_ + i] + bh[2 * H_ + i];
            const float r = 1.f / (1.f + expf(-(xr + hr)));
            const float z = 1.f / (1.f + expf(-(xz + hz)));
            const float n = tanhf(xn + r * hn);
            const float hNew = (1.f - z) * n + z * hf[b * H_ + i];
            hf[b * H_ + i] = hNew;
            const __bf16 hb = (__bf16)hNew;
            hbf[b * H_ + i] = hb;
            hs[((size_t)t * B_ + b) * H_ + i] = hb;
        }
        gridSync(barCnt, barPhase, NWG_, inCluster);
    }
}

// =====================================================================
// Output MLP: one 16x16 tile per wave, bf16 WMMA, fp32 accumulate.
// K-loop software-pipelined (prefetch next fragments during WMMA).
// =====================================================================
__global__ void __launch_bounds__(256) kMlp1(const __bf16* __restrict__ hs,
                                             const __bf16* __restrict__ w1b,
                                             const float* __restrict__ b1,
                                             __bf16* __restrict__ y)
{
    const int lane = threadIdx.x & 31, wave = threadIdx.x >> 5;
    const int NTI = H_ / 16;                                   // 56 N tiles
    const int task = blockIdx.x * 8 + wave;
    if (task >= (ROWS_ / 16) * NTI) return;
    const int m = task / NTI, n = task % NTI;
    const int row   = m * 16 + (lane & 15);
    const int khalf = (lane >> 4) * 16;
    const int jrow  = n * 16 + (lane & 15);
    const __bf16* aBase = hs  + (size_t)row * H_ + khalf;
    const __bf16* bBase = w1b + (size_t)jrow * H_ + (lane >> 4) * 16;
    v8f acc = {0.f,0.f,0.f,0.f,0.f,0.f,0.f,0.f};
    v8bf lo = *(const v8bf*)(aBase);
    v8bf hi = *(const v8bf*)(aBase + 8);
    v16bf bfrag = *(const v16bf*)(bBase);
    for (int kt = 0; kt < KT_; ++kt) {
        const int kn = ((kt + 1 < KT_) ? kt + 1 : kt) * 32;
        const v8bf loN = *(const v8bf*)(aBase + kn);
        const v8bf hiN = *(const v8bf*)(aBase + kn + 8);
        const v16bf bfragN = *(const v16bf*)(bBase + kn);
        const v16bf afrag = __builtin_shufflevector(lo, hi, 0, 1, 2, 3, 4, 5, 6, 7,
                                                            8, 9, 10, 11, 12, 13, 14, 15);
        acc = __builtin_amdgcn_wmma_f32_16x16x32_bf16(
            false, afrag, false, bfrag, (short)0, acc, false, false);
        lo = loN; hi = hiN; bfrag = bfragN;
    }
    const int col = n * 16 + (lane & 15);
    const float bias = b1[col];
    #pragma unroll
    for (int v = 0; v < 8; ++v) {
        const int r = m * 16 + v + (lane >> 4) * 8;
        float val = acc[v] + bias;
        val = val > 0.f ? val : 0.f;                           // relu
        y[(size_t)r * H_ + col] = (__bf16)val;
    }
}

__global__ void __launch_bounds__(256) kMlp2(const __bf16* __restrict__ y,
                                             const __bf16* __restrict__ w2b,
                                             const float* __restrict__ b2,
                                             float* __restrict__ logits)
{
    const int lane = threadIdx.x & 31, wave = threadIdx.x >> 5;
    const int NTI = A_ / 16;                                   // 16 N tiles
    const int task = blockIdx.x * 8 + wave;
    if (task >= (ROWS_ / 16) * NTI) return;
    const int m = task / NTI, n = task % NTI;
    const int row   = m * 16 + (lane & 15);
    const int khalf = (lane >> 4) * 16;
    const int jrow  = n * 16 + (lane & 15);
    const __bf16* aBase = y   + (size_t)row * H_ + khalf;
    const __bf16* bBase = w2b + (size_t)jrow * H_ + (lane >> 4) * 16;
    v8f acc = {0.f,0.f,0.f,0.f,0.f,0.f,0.f,0.f};
    v8bf lo = *(const v8bf*)(aBase);
    v8bf hi = *(const v8bf*)(aBase + 8);
    v16bf bfrag = *(const v16bf*)(bBase);
    for (int kt = 0; kt < KT_; ++kt) {
        const int kn = ((kt + 1 < KT_) ? kt + 1 : kt) * 32;
        const v8bf loN = *(const v8bf*)(aBase + kn);
        const v8bf hiN = *(const v8bf*)(aBase + kn + 8);
        const v16bf bfragN = *(const v16bf*)(bBase + kn);
        const v16bf afrag = __builtin_shufflevector(lo, hi, 0, 1, 2, 3, 4, 5, 6, 7,
                                                            8, 9, 10, 11, 12, 13, 14, 15);
        acc = __builtin_amdgcn_wmma_f32_16x16x32_bf16(
            false, afrag, false, bfrag, (short)0, acc, false, false);
        lo = loN; hi = hiN; bfrag = bfragN;
    }
    const int col = n * 16 + (lane & 15);
    const float bias = b2[col];
    #pragma unroll
    for (int v = 0; v < 8; ++v) {
        const int r = m * 16 + v + (lane >> 4) * 8;
        logits[(size_t)r * A_ + col] = acc[v] + bias;
    }
}

// log-softmax over A=256 per (t,b) row, write transposed [B,A,T] fp32
__global__ void __launch_bounds__(256) kLogSoftmax(const float* __restrict__ logits,
                                                   float* __restrict__ out)
{
    const int lane = threadIdx.x & 31, wave = threadIdx.x >> 5;
    const int rw = blockIdx.x * 8 + wave;
    if (rw >= ROWS_) return;
    const float* lp = logits + (size_t)rw * A_;
    float vals[8];
    float mx = -3.4e38f;
    #pragma unroll
    for (int e = 0; e < 8; ++e) { vals[e] = lp[lane * 8 + e]; mx = fmaxf(mx, vals[e]); }
    #pragma unroll
    for (int off = 16; off; off >>= 1) mx = fmaxf(mx, __shfl_xor(mx, off, 32));
    float s = 0.f;
    #pragma unroll
    for (int e = 0; e < 8; ++e) s += expf(vals[e] - mx);
    #pragma unroll
    for (int off = 16; off; off >>= 1) s += __shfl_xor(s, off, 32);
    const float lse = mx + logf(s);
    const int t = rw / B_, b = rw % B_;                        // row = t*B + b
    #pragma unroll
    for (int e = 0; e < 8; ++e) {
        const int a = lane * 8 + e;
        out[((size_t)b * A_ + a) * T_ + t] = vals[e] - lse;
    }
}

// =====================================================================
extern "C" void kernel_launch(void* const* d_in, const int* in_sizes, int n_in,
                              void* d_out, int out_size, void* d_ws, size_t ws_size,
                              hipStream_t stream)
{
    const float* lf   = (const float*)d_in[0];
    const float* gold = (const float*)d_in[1];
    const float* pw   = (const float*)d_in[2];
    const float* pb   = (const float*)d_in[3];
    const float* upw  = (const float*)d_in[4];
    const float* upb  = (const float*)d_in[5];
    const float* ew   = (const float*)d_in[6];
    const float* eb   = (const float*)d_in[7];
    const float* wi   = (const float*)d_in[8];
    const float* wh   = (const float*)d_in[9];
    const float* bi   = (const float*)d_in[10];
    const float* bh   = (const float*)d_in[11];
    const float* w1   = (const float*)d_in[12];
    const float* b1   = (const float*)d_in[13];
    const float* w2   = (const float*)d_in[14];
    const float* b2   = (const float*)d_in[15];
    float* out = (float*)d_out;

    char* ws = (char*)d_ws;
    size_t off = 0;
    auto take = [&](size_t bytes) {
        char* p = ws + off;
        off = (off + bytes + 255) & ~(size_t)255;
        return p;
    };
    float*    x1     = (float*)   take((size_t)B_ * C_ * L_ * 4);
    float*    cond   = (float*)   take((size_t)SEG_ * B_ * C_ * 4);
    float*    U      = (float*)   take((size_t)SEG_ * B_ * H3_ * 4);
    float*    vvec   = (float*)   take((size_t)H3_ * 4);
    float*    cvec   = (float*)   take((size_t)H3_ * 4);
    float*    gh     = (float*)   take((size_t)B_ * H3_ * 4);
    float*    hf     = (float*)   take((size_t)B_ * H_ * 4);
    __bf16*   hbf    = (__bf16*)  take((size_t)MPAD_ * H_ * 2);   // 16 zero-padded rows
    __bf16*   hs     = (__bf16*)  take((size_t)ROWS_ * H_ * 2);
    __bf16*   ybuf   = (__bf16*)  take((size_t)ROWS_ * H_ * 2);
    float*    logits = (float*)   take((size_t)ROWS_ * A_ * 4);
    __bf16*   whp    = (__bf16*)  take((size_t)NWG_ * NT_ * KT_ * 32 * 16 * 2);
    __bf16*   w1b    = (__bf16*)  take((size_t)H_ * H_ * 2);
    __bf16*   w2b    = (__bf16*)  take((size_t)A_ * H_ * 2);
    unsigned* bar    = (unsigned*)take(256);
    (void)in_sizes; (void)n_in; (void)out_size; (void)ws_size;

    // ---- cond path + algebraic GRU-input precompute ----
    kProj<<<(B_ * C_ * L_ + 255) / 256, 256, 0, stream>>>(lf, pw, pb, x1);
    kUp<<<(SEG_ * B_ * C_ + 255) / 256, 256, 0, stream>>>(x1, upw, upb, cond);
    kVC<<<(H3_ + 255) / 256, 256, 0, stream>>>(wi, ew, eb, bi, vvec, cvec);
    kCondProj<<<(SEG_ * B_ * H3_ + 255) / 256, 256, 0, stream>>>(cond, wi, U);

    // ---- weight packing (bf16 / WMMA fragments) + state init ----
    kPackWh<<<(NWG_ * NT_ * KT_ * 32 + 255) / 256, 256, 0, stream>>>(wh, whp);
    kCvtBf16<<<(H_ * H_ + 255) / 256, 256, 0, stream>>>(w1, w1b, H_ * H_);
    kCvtBf16<<<(A_ * H_ + 255) / 256, 256, 0, stream>>>(w2, w2b, A_ * H_);
    kZeroState<<<1, 256, 0, stream>>>(hf, hbf, bar);

    // ---- persistent cluster GRU: 16 WGs, LDS-resident Wh, per-step sync ----
    const size_t ldsBytes = (size_t)NT_ * KT_ * 32 * 16 * 2;   // 315,392 B < 320 KB
    kGru<<<NWG_, 128, ldsBytes, stream>>>(whp, U, vvec, cvec, gold, bh,
                                          gh, hf, hbf, hs, bar, bar + 1);

    // ---- output MLP + log-softmax ----
    kMlp1<<<(ROWS_ / 16) * (H_ / 16) / 8, 256, 0, stream>>>(hs, w1b, b1, ybuf);
    kMlp2<<<(ROWS_ / 16) * (A_ / 16) / 8, 256, 0, stream>>>(ybuf, w2b, b2, logits);
    kLogSoftmax<<<(ROWS_ + 7) / 8, 256, 0, stream>>>(logits, out);
}
